// _WindowAttnV2_64965675319858
// MI455X (gfx1250) — matrix operand order repaired and checked by reading, not compiled, not run on hardware
//
#include <hip/hip_runtime.h>
#include <hip/hip_bf16.h>

typedef __attribute__((ext_vector_type(16))) _Float16 v16h;
typedef __attribute__((ext_vector_type(8)))  _Float16 v8h;
typedef __attribute__((ext_vector_type(8)))  float    v8f;
typedef __attribute__((ext_vector_type(4)))  int      v4i;

#define NH   8
#define NTOK 256
#define HD   32
#define DIM  256
#define BW   256
#define NROW (BW * NTOK)     // 65536

// ---------------------------------------------------------------------------
// gfx1250 async global->LDS copy (ASYNCcnt-tracked), with sync fallback.
// Builtin parameter types (from hipcc diagnostic): v4i addrspace(1)* src,
// v4i addrspace(3)* lds, imm offset, imm cpol.
// ---------------------------------------------------------------------------
#if defined(__AMDGCN__) && __has_builtin(__builtin_amdgcn_global_load_async_to_lds_b128)
#define ASYNC_LDS 1
typedef v4i __attribute__((address_space(1)))* gas_v4i_ptr;
typedef v4i __attribute__((address_space(3)))* lds_v4i_ptr;
#else
#define ASYNC_LDS 0
#endif

__device__ __forceinline__ void async_copy_b128(const _Float16* g, _Float16* l) {
#if ASYNC_LDS
  __builtin_amdgcn_global_load_async_to_lds_b128(
      (gas_v4i_ptr)g, (lds_v4i_ptr)l, 0, 0);
#else
  *(v8h*)l = *(const v8h*)g;
#endif
}

__device__ __forceinline__ void wait_async_lds() {
#if ASYNC_LDS
#if __has_builtin(__builtin_amdgcn_s_wait_asynccnt)
  __builtin_amdgcn_s_wait_asynccnt(0);
#else
  asm volatile("s_wait_asynccnt 0" ::: "memory");
#endif
#endif
}

// ---------------------------------------------------------------------------
// WMMA helpers (gfx1250 wave32): D = A(16x32 f16) * B(32x16 f16) + C(f32)
// A frag: lanes 0-15 row M=lane, halves[0..7]=K0..7, halves[8..15]=K8..15
//         lanes 16-31 same rows,  halves[0..7]=K16..23, halves[8..15]=K24..31
// B frag: lanes 0-15 col N=lane holds K0..15; lanes 16-31 col N=lane-16 hold K16..31
// C/D:    vgpr i, lanes0-15 -> (M=i, N=lane); lanes16-31 -> (M=i+8, N=lane-16)
// ---------------------------------------------------------------------------
__device__ __forceinline__ v8f wmma_f16(v16h a, v16h b, v8f c) {
  return __builtin_amdgcn_wmma_f32_16x16x32_f16(false, a, false, b, (short)0, c,
                                                false, false);
}

__device__ __forceinline__ v16h load_afrag(const _Float16* rowbase, int lane) {
  int k0 = (lane & 16);                            // 0 or 16 halves
  v8h lo = *(const v8h*)(rowbase + k0);            // K k0..k0+7
  v8h hi = *(const v8h*)(rowbase + 8 + k0);        // K k0+8..k0+15
  v16h a;
#pragma unroll
  for (int i = 0; i < 8; ++i) { a[i] = lo[i]; a[8 + i] = hi[i]; }
  return a;
}

__device__ __forceinline__ v16h load_frag16(const _Float16* p) {
  v8h lo = *(const v8h*)(p);
  v8h hi = *(const v8h*)(p + 8);
  v16h b;
#pragma unroll
  for (int i = 0; i < 8; ++i) { b[i] = lo[i]; b[8 + i] = hi[i]; }
  return b;
}

// ---------------------------------------------------------------------------
// Elementwise f32 -> f16 cast
// ---------------------------------------------------------------------------
__global__ void cast_f32_f16(const float* __restrict__ src,
                             _Float16* __restrict__ dst, int n) {
  for (int i = blockIdx.x * blockDim.x + threadIdx.x; i < n;
       i += gridDim.x * blockDim.x)
    dst[i] = (_Float16)src[i];
}

// ---------------------------------------------------------------------------
// Continuous position-bias MLP: relu(table @ w1^T + b1) @ w2^T -> 16*sigmoid
// ---------------------------------------------------------------------------
__global__ void cpb_mlp(const float* __restrict__ table,
                        const float* __restrict__ w1,
                        const float* __restrict__ b1,
                        const float* __restrict__ w2,
                        float* __restrict__ rpb) {
  int i = blockIdx.x * blockDim.x + threadIdx.x;
  if (i >= 961) return;
  float t0 = table[i * 2 + 0], t1 = table[i * 2 + 1];
  float s[NH] = {};
  for (int j = 0; j < 512; ++j) {
    float h = t0 * w1[j * 2 + 0] + t1 * w1[j * 2 + 1] + b1[j];
    h = fmaxf(h, 0.f);
#pragma unroll
    for (int n = 0; n < NH; ++n) s[n] += h * w2[n * 512 + j];
  }
#pragma unroll
  for (int n = 0; n < NH; ++n)
    rpb[i * NH + n] = 16.f / (1.f + __expf(-s[n]));
}

// bias_full[h][n][m] = rpb[rel_index[n][m]][h]
__global__ void bias_gather(const float* __restrict__ rpb,
                            const int* __restrict__ rel_index,
                            float* __restrict__ bias_full) {
  int idx = blockIdx.x * blockDim.x + threadIdx.x;  // over 8*65536
  if (idx >= NH * NTOK * NTOK) return;
  int h = idx >> 16, nm = idx & 65535;
  bias_full[idx] = rpb[rel_index[nm] * NH + h];
}

// ---------------------------------------------------------------------------
// Shared GEMM tile machinery: block = 256 thr (8 waves), tile 128 x 64,
// K chunked by 32, double-buffered LDS filled with async global->LDS copies.
// ---------------------------------------------------------------------------
__device__ __forceinline__ void stage_chunk(const _Float16* __restrict__ A,
                                            const _Float16* __restrict__ B,
                                            int r0, int c0, int k0,
                                            _Float16* Abuf, _Float16* Bbuf,
                                            int t) {
#pragma unroll
  for (int i = 0; i < 2; ++i) {                 // A: 128x32 halves = 512 xfers
    int idx = t * 2 + i, row = idx >> 2, seg = idx & 3;
    async_copy_b128(A + (size_t)(r0 + row) * DIM + k0 + seg * 8,
                    Abuf + row * 32 + seg * 8);
  }
  {                                             // B: 64x32 halves = 256 xfers
    int row = t >> 2, seg = t & 3;
    async_copy_b128(B + (size_t)(c0 + row) * DIM + k0 + seg * 8,
                    Bbuf + row * 32 + seg * 8);
  }
}

// ---------------------------------------------------------------------------
// QKV GEMM: out[r][c] = sum_k xh[r][k]*wh[c][k] + qkv_bias[c]
// write layout: qkv[(sec*BW+b)*NH+h][tok][d]  f16  (sec 0=q,1=k,2=v)
// ---------------------------------------------------------------------------
#define QKV_SEC ((size_t)BW * NH * NTOK * HD)

__global__ __launch_bounds__(256) void qkv_gemm(
    const _Float16* __restrict__ xh, const _Float16* __restrict__ wh,
    const float* __restrict__ q_bias, const float* __restrict__ v_bias,
    _Float16* __restrict__ qkv) {
  __shared__ __align__(16) _Float16 As[2][128 * 32];
  __shared__ __align__(16) _Float16 Bs[2][64 * 32];
  int t = threadIdx.x, lane = t & 31, w = t >> 5;
  int r0 = blockIdx.x * 128, c0 = blockIdx.y * 64;
  v8f acc[4] = {};

  stage_chunk(xh, wh, r0, c0, 0, As[0], Bs[0], t);
#pragma unroll 1
  for (int c = 0; c < 8; ++c) {
    wait_async_lds();
    __syncthreads();
    if (c + 1 < 8)
      stage_chunk(xh, wh, r0, c0, (c + 1) * 32, As[(c + 1) & 1], Bs[(c + 1) & 1], t);
    const _Float16* Ab = As[c & 1];
    const _Float16* Bb = Bs[c & 1];
    v16h a = load_afrag(Ab + (w * 16 + (lane & 15)) * 32, lane);
#pragma unroll
    for (int ct = 0; ct < 4; ++ct) {
      v16h b = load_frag16(Bb + (ct * 16 + (lane & 15)) * 32 + (lane & 16));
      acc[ct] = wmma_f16(a, b, acc[ct]);
    }
  }

  int rbase = r0 + w * 16 + ((lane >> 4) << 3);
#pragma unroll
  for (int ct = 0; ct < 4; ++ct) {
#pragma unroll
    for (int i = 0; i < 8; ++i) {
      int r = rbase + i;
      int cc = c0 + ct * 16 + (lane & 15);
      int sec = cc >> 8, ch = cc & 255;
      float bias = (sec == 0) ? q_bias[ch] : (sec == 2 ? v_bias[ch] : 0.f);
      int b_ = r >> 8, tok = r & 255, h = ch >> 5, d = ch & 31;
      qkv[(((size_t)sec * BW + b_) * NH + h) * NTOK * HD + tok * HD + d] =
          (_Float16)(acc[ct][i] + bias);
    }
  }
}

// ---------------------------------------------------------------------------
// Fused window attention: one block per (window b, head h).
// S = (q_n * scale) k_n^T ; softmax(S + bias) ; O = P v
// Each wave owns a 16x256 strip of S entirely in registers.
// ---------------------------------------------------------------------------
__global__ __launch_bounds__(256) void attn_fused(
    const _Float16* __restrict__ qkv, const float* __restrict__ logit_scale,
    const float* __restrict__ bias_full, _Float16* __restrict__ attn_out) {
  __shared__ __align__(16) _Float16 qs[NTOK * HD];    // [tok][d] normalized*scale
  __shared__ __align__(16) _Float16 ks[NTOK * HD];    // [tok][d] normalized
  __shared__ __align__(16) _Float16 vt[HD * NTOK];    // [d][tok] (transposed)
  __shared__ __align__(16) _Float16 ps[8][16 * 32];   // wave-private P staging

  int t = threadIdx.x, lane = t & 31, w = t >> 5;
  int b = blockIdx.x >> 3, h = blockIdx.x & 7;
  size_t base = ((size_t)b * NH + h) * NTOK * HD;
  const _Float16* qg = qkv + base;
  const _Float16* kg = qkv + QKV_SEC + base;
  const _Float16* vg = qkv + 2 * QKV_SEC + base;
  float scale = __expf(fminf(logit_scale[h], 4.6051701859880914f));

  {  // stage: one token per thread
    int tok = t;
    float qb[HD], kb[HD];
    float sq = 0.f, sk = 0.f;
#pragma unroll
    for (int d = 0; d < HD; ++d) {
      qb[d] = (float)qg[tok * HD + d]; sq += qb[d] * qb[d];
      kb[d] = (float)kg[tok * HD + d]; sk += kb[d] * kb[d];
    }
    float iq = scale / fmaxf(sqrtf(sq), 1e-12f);
    float ik = 1.f / fmaxf(sqrtf(sk), 1e-12f);
#pragma unroll
    for (int d = 0; d < HD; ++d) {
      qs[tok * HD + d] = (_Float16)(qb[d] * iq);
      ks[tok * HD + d] = (_Float16)(kb[d] * ik);
      vt[d * NTOK + tok] = vg[tok * HD + d];
    }
  }
  __syncthreads();

  // ---- S = q k^T : 16 tiles, K = hd = 32 = one WMMA each ----
  v8f acc[16];
  v16h aq = load_afrag(qs + (w * 16 + (lane & 15)) * HD, lane);
#pragma unroll
  for (int ct = 0; ct < 16; ++ct) {
    v16h bk = load_frag16(ks + (ct * 16 + (lane & 15)) * HD + (lane & 16));
    v8f z = {};
    acc[ct] = wmma_f16(aq, bk, z);
  }

  // ---- bias + softmax over 256 cols (rows live in this wave) ----
  const float* bf = bias_full + (size_t)h * NTOK * NTOK;
  int rbase = w * 16 + ((lane >> 4) << 3);
  float mrow[8], srow[8];
#pragma unroll
  for (int i = 0; i < 8; ++i) mrow[i] = -3.0e38f;
#pragma unroll
  for (int ct = 0; ct < 16; ++ct)
#pragma unroll
    for (int i = 0; i < 8; ++i) {
      float v = acc[ct][i] + bf[(rbase + i) * NTOK + ct * 16 + (lane & 15)];
      acc[ct][i] = v;
      mrow[i] = fmaxf(mrow[i], v);
    }
#pragma unroll
  for (int i = 0; i < 8; ++i) {
#pragma unroll
    for (int mask = 1; mask < 16; mask <<= 1)
      mrow[i] = fmaxf(mrow[i], __shfl_xor(mrow[i], mask, 16));
    srow[i] = 0.f;
  }
#pragma unroll
  for (int ct = 0; ct < 16; ++ct)
#pragma unroll
    for (int i = 0; i < 8; ++i) {
      float e = __expf(acc[ct][i] - mrow[i]);
      acc[ct][i] = e;
      srow[i] += e;
    }
#pragma unroll
  for (int i = 0; i < 8; ++i) {
#pragma unroll
    for (int mask = 1; mask < 16; mask <<= 1)
      srow[i] += __shfl_xor(srow[i], mask, 16);
    srow[i] = 1.f / srow[i];
  }
#pragma unroll
  for (int ct = 0; ct < 16; ++ct)
#pragma unroll
    for (int i = 0; i < 8; ++i) acc[ct][i] *= srow[i];

  // ---- O = P v : loop K(keys) in chunks of 32 via wave-private transpose ----
  v8f oacc[2] = {};
  _Float16* pw = ps[w];
#pragma unroll 1
  for (int kt2 = 0; kt2 < 8; ++kt2) {
#pragma unroll
    for (int half = 0; half < 2; ++half) {
      int ct = kt2 * 2 + half;
#pragma unroll
      for (int i = 0; i < 8; ++i)
        pw[(i + ((lane >> 4) << 3)) * 32 + half * 16 + (lane & 15)] =
            (_Float16)acc[ct][i];
    }
    asm volatile("s_wait_dscnt 0" ::: "memory");   // DS RAW within wave
    v16h pa = load_afrag(pw + (lane & 15) * 32, lane);
#pragma unroll
    for (int ot = 0; ot < 2; ++ot) {
      v16h bv = load_frag16(vt + (ot * 16 + (lane & 15)) * NTOK + kt2 * 32 +
                            (lane & 16));
      oacc[ot] = wmma_f16(pa, bv, oacc[ot]);
    }
    asm volatile("s_wait_dscnt 0" ::: "memory");   // WAR before next store
  }
#pragma unroll
  for (int ot = 0; ot < 2; ++ot)
#pragma unroll
    for (int i = 0; i < 8; ++i) {
      int tok = rbase + i, d = ot * 16 + (lane & 15);
      attn_out[((size_t)b * NTOK + tok) * DIM + h * HD + d] =
          (_Float16)oacc[ot][i];
    }
}

// ---------------------------------------------------------------------------
// Proj GEMM: out[r][c] = sum_k ah[r][k]*wh[c][k] + pb[c]   (f32 output)
// ---------------------------------------------------------------------------
__global__ __launch_bounds__(256) void proj_gemm(
    const _Float16* __restrict__ ah, const _Float16* __restrict__ wh,
    const float* __restrict__ pb, float* __restrict__ out) {
  __shared__ __align__(16) _Float16 As[2][128 * 32];
  __shared__ __align__(16) _Float16 Bs[2][64 * 32];
  int t = threadIdx.x, lane = t & 31, w = t >> 5;
  int r0 = blockIdx.x * 128, c0 = blockIdx.y * 64;
  v8f acc[4] = {};

  stage_chunk(ah, wh, r0, c0, 0, As[0], Bs[0], t);
#pragma unroll 1
  for (int c = 0; c < 8; ++c) {
    wait_async_lds();
    __syncthreads();
    if (c + 1 < 8)
      stage_chunk(ah, wh, r0, c0, (c + 1) * 32, As[(c + 1) & 1], Bs[(c + 1) & 1], t);
    const _Float16* Ab = As[c & 1];
    const _Float16* Bb = Bs[c & 1];
    v16h a = load_afrag(Ab + (w * 16 + (lane & 15)) * 32, lane);
#pragma unroll
    for (int ct = 0; ct < 4; ++ct) {
      v16h b = load_frag16(Bb + (ct * 16 + (lane & 15)) * 32 + (lane & 16));
      acc[ct] = wmma_f16(a, b, acc[ct]);
    }
  }

  int rbase = r0 + w * 16 + ((lane >> 4) << 3);
#pragma unroll
  for (int ct = 0; ct < 4; ++ct)
#pragma unroll
    for (int i = 0; i < 8; ++i) {
      int cc = c0 + ct * 16 + (lane & 15);
      out[(size_t)(rbase + i) * DIM + cc] = acc[ct][i] + pb[cc];
    }
}

// ---------------------------------------------------------------------------
extern "C" void kernel_launch(void* const* d_in, const int* in_sizes, int n_in,
                              void* d_out, int out_size, void* d_ws,
                              size_t ws_size, hipStream_t stream) {
  const float* x        = (const float*)d_in[0];   // [256,256,256]
  const float* qkv_w    = (const float*)d_in[1];   // [768,256]
  const float* q_bias   = (const float*)d_in[2];   // [256]
  const float* v_bias   = (const float*)d_in[3];   // [256]
  const float* lscale   = (const float*)d_in[4];   // [8]
  const float* cpb_w1   = (const float*)d_in[5];   // [512,2]
  const float* cpb_b1   = (const float*)d_in[6];   // [512]
  const float* cpb_w2   = (const float*)d_in[7];   // [8,512]
  const float* proj_w   = (const float*)d_in[8];   // [256,256]
  const float* proj_b   = (const float*)d_in[9];   // [256]
  const float* ctable   = (const float*)d_in[10];  // [961,2]
  const int*   rel_idx  = (const int*)d_in[11];    // [256,256]
  float* out = (float*)d_out;

  // workspace layout (bytes, 256-aligned sections)
  char* ws = (char*)d_ws;
  _Float16* xh     = (_Float16*)(ws);                         // 33,554,432
  _Float16* qkvwh  = (_Float16*)(ws + 33554432);              //    393,216
  _Float16* projwh = (_Float16*)(ws + 33554432 + 393216);     //    131,072
  _Float16* qkvh   = (_Float16*)(ws + 34078720);              // 100,663,296
  _Float16* attnh  = (_Float16*)(ws + 134742016);             // 33,554,432
  float*    rpb    = (float*)   (ws + 168296448);             //     32,768
  float*    biasf  = (float*)   (ws + 168329216);             //  2,097,152

  cast_f32_f16<<<4096, 256, 0, stream>>>(x, xh, NROW * DIM);
  cast_f32_f16<<<768, 256, 0, stream>>>(qkv_w, qkvwh, 3 * DIM * DIM);
  cast_f32_f16<<<256, 256, 0, stream>>>(proj_w, projwh, DIM * DIM);

  cpb_mlp<<<4, 256, 0, stream>>>(ctable, cpb_w1, cpb_b1, cpb_w2, rpb);
  bias_gather<<<(NH * NTOK * NTOK + 255) / 256, 256, 0, stream>>>(rpb, rel_idx,
                                                                  biasf);

  qkv_gemm<<<dim3(NROW / 128, (3 * DIM) / 64), 256, 0, stream>>>(
      xh, qkvwh, q_bias, v_bias, qkvh);

  attn_fused<<<BW * NH, 256, 0, stream>>>(qkvh, lscale, biasf, attnh);

  proj_gemm<<<dim3(NROW / 128, DIM / 64), 256, 0, stream>>>(attnh, projwh,
                                                            proj_b, out);
}
